// TokenAgentAttention_12240656793970
// MI455X (gfx1250) — matrix operand order repaired and checked by reading, not compiled, not run on hardware
//
#include <hip/hip_runtime.h>
#include <hip/hip_bf16.h>

typedef __attribute__((ext_vector_type(16))) _Float16 v16h;
typedef __attribute__((ext_vector_type(8)))  _Float16 v8h;
typedef __attribute__((ext_vector_type(8)))  float    v8f;

#define NB     128
#define SEQ    197
#define DIMC   768
#define NHEAD  12
#define HDIM   64
#define NAGENT 40
#define WINS   14
#define MTOK   (NB*SEQ)          // 25216
#define SCALE  0.125f            // 64^-0.5

// ---------------------------------------------------------------------------
// fragment packers (contiguous k = 16*half + j order, same for A and B, wave32)
// ---------------------------------------------------------------------------
__device__ __forceinline__ v16h pack_a_h(const _Float16* row, int k0, int half) {
  // A 16x32 f16: lane holds K = k0+half*16+j  and  k0+half*16+8+j
  v8h lo = *(const v8h*)(row + k0 + half*16);
  v8h hi = *(const v8h*)(row + k0 + half*16 + 8);
  v16h a;
  for (int j = 0; j < 8; ++j) { a[j] = lo[j]; a[8+j] = hi[j]; }
  return a;
}
__device__ __forceinline__ v16h pack_a_f32(const float* row, int k0, int half) {
  v16h a;
  for (int j = 0; j < 8; ++j) {
    a[j]   = (_Float16)row[k0 + half*16 + j];
    a[8+j] = (_Float16)row[k0 + half*16 + 8 + j];
  }
  return a;
}
__device__ __forceinline__ v16h pack_a_f32_guard(const float* row, int k0, int half, int kmax) {
  v16h a;
  for (int j = 0; j < 8; ++j) {
    int ka = k0 + half*16 + j, kb = k0 + half*16 + 8 + j;
    a[j]   = (ka < kmax) ? (_Float16)row[ka] : (_Float16)0.f;
    a[8+j] = (kb < kmax) ? (_Float16)row[kb] : (_Float16)0.f;
  }
  return a;
}
__device__ __forceinline__ v16h zero_frag() {
  v16h z;
  for (int j = 0; j < 16; ++j) z[j] = (_Float16)0.f;
  return z;
}
__device__ __forceinline__ v8f wmma16(v16h a, v16h b, v8f c) {
  return __builtin_amdgcn_wmma_f32_16x16x32_f16(false, a, false, b, (short)0, c, false, false);
}

// ---------------------------------------------------------------------------
// CDNA5 async staging: GLOBAL_LOAD_ASYNC_TO_LDS_B128 (ASYNCcnt tracked).
// half_idx is the half-element offset inside the (single, offset-0) shared
// array. Invalid rows are zero-filled via ds_store (DScnt, covered by barrier).
// ---------------------------------------------------------------------------
__device__ __forceinline__ void stage16_async(_Float16* sh_generic, unsigned half_idx,
                                              const _Float16* gsrc, bool valid) {
  if (valid) {
    unsigned lds_off = half_idx * 2u;                 // byte offset in LDS
    unsigned long long ga = (unsigned long long)(const void*)gsrc;
    asm volatile("global_load_async_to_lds_b128 %0, %1, off"
                 :: "v"(lds_off), "v"(ga) : "memory");
  } else {
    v8h z; for (int j = 0; j < 8; ++j) z[j] = (_Float16)0.f;
    *(v8h*)(sh_generic + half_idx) = z;
  }
}
__device__ __forceinline__ void wait_async0() {
  asm volatile("s_wait_asynccnt 0x0" ::: "memory");
}

// ---------------------------------------------------------------------------
// elementwise f32 -> f16 cast
// ---------------------------------------------------------------------------
__global__ void cast_f16_kernel(const float* __restrict__ in,
                                _Float16* __restrict__ out, long long n) {
  long long i = (long long)blockIdx.x * blockDim.x + threadIdx.x;
  if (i < n) out[i] = (_Float16)in[i];
}

// ---------------------------------------------------------------------------
// GEMM: C[M][N] = A[M][K] * B[N][K]^T (+bias[n]); f16 in, f32 accumulate.
// block tile 128x128, 8 waves of 32x64 (2x4 WMMA tiles), K-step 32.
// Double-buffered LDS staged with async-to-LDS loads.
// ---------------------------------------------------------------------------
template <bool OUT_F16>
__global__ void __launch_bounds__(256)
gemm_nt_wmma(const _Float16* __restrict__ A, const _Float16* __restrict__ Bm,
             const float* __restrict__ bias,
             float* __restrict__ Cf, _Float16* __restrict__ Ch,
             int M, int Nn, int K) {
  constexpr int BM = 128, BN = 128, BK = 32;
  constexpr unsigned BUF = (BM + BN) * BK;          // 8192 halves / 16 KB
  __shared__ _Float16 sh[2 * BUF];                  // 32 KB, LDS offset 0

  const int tid  = threadIdx.x;
  const int lane = tid & 31, wid = tid >> 5;
  const int wm = wid >> 1, wn = wid & 1;            // 4 x 2 wave grid
  const int half = lane >> 4, lr = lane & 15;
  const int mblk = blockIdx.x * BM;
  const int nblk = blockIdx.y * BN;

  v8f acc[2][4];
  for (int i = 0; i < 2; ++i)
    for (int j = 0; j < 4; ++j)
      for (int r = 0; r < 8; ++r) acc[i][j][r] = 0.f;

  // stage one K-tile (A: 128x32, B: 128x32) into buffer p
  auto stage_tiles = [&](int p, int k0) {
    unsigned base = (unsigned)p * BUF;
    for (int it = 0; it < 2; ++it) {                // A: 512 x 16B
      int t = tid + it * 256;
      int r = t >> 2, cb = (t & 3) * 8;
      stage16_async(sh, base + (unsigned)(r * BK + cb),
                    A + (size_t)(mblk + r) * K + k0 + cb, (mblk + r) < M);
    }
    for (int it = 0; it < 2; ++it) {                // B: 512 x 16B
      int t = tid + it * 256;
      int r = t >> 2, cb = (t & 3) * 8;
      stage16_async(sh, base + (unsigned)(BM * BK) + (unsigned)(r * BK + cb),
                    Bm + (size_t)(nblk + r) * K + k0 + cb, (nblk + r) < Nn);
    }
  };

  int p = 0;
  stage_tiles(p, 0);
  wait_async0();
  __syncthreads();

  for (int k0 = 0; k0 < K; k0 += BK) {
    int np = p ^ 1;
    if (k0 + BK < K) stage_tiles(np, k0 + BK);      // overlap with compute

    const _Float16* shA = sh + (unsigned)p * BUF;
    const _Float16* shB = shA + BM * BK;

    v16h af[2], bf[4];
    for (int tm = 0; tm < 2; ++tm)
      af[tm] = pack_a_h(shA + (wm*32 + tm*16 + lr) * BK, 0, half);
    for (int tn = 0; tn < 4; ++tn)
      bf[tn] = *(const v16h*)(shB + (wn*64 + tn*16 + lr) * BK + half * 16);

    for (int tm = 0; tm < 2; ++tm)
      for (int tn = 0; tn < 4; ++tn)
        acc[tm][tn] = wmma16(af[tm], bf[tn], acc[tm][tn]);

    wait_async0();
    __syncthreads();
    p = np;
  }

  for (int tm = 0; tm < 2; ++tm)
    for (int tn = 0; tn < 4; ++tn) {
      int col = nblk + wn*64 + tn*16 + lr;
      if (col >= Nn) continue;
      float bv = bias ? bias[col] : 0.f;
      for (int r = 0; r < 8; ++r) {
        int row = mblk + wm*32 + tm*16 + half*8 + r;
        if (row >= M) continue;
        float v = acc[tm][tn][r] + bv;
        if constexpr (OUT_F16) Ch[(size_t)row * Nn + col] = (_Float16)v;
        else                   Cf[(size_t)row * Nn + col] = v;
      }
    }
}

// ---------------------------------------------------------------------------
// positional bias precompute: pos[12][40][197], ab[12][197][40]
// bilinear 7->14 (half-pixel centers, like jax.image.resize 'bilinear')
// ---------------------------------------------------------------------------
__device__ __forceinline__ float bilerp7(const float* t7, int y, int x) {
  float fy = (y + 0.5f) * 0.5f - 0.5f;
  float fx = (x + 0.5f) * 0.5f - 0.5f;
  int y0 = (int)floorf(fy), x0 = (int)floorf(fx);
  float wy = fy - (float)y0, wx = fx - (float)x0;
  int y0c = y0 < 0 ? 0 : (y0 > 6 ? 6 : y0);
  int y1 = y0 + 1; int y1c = y1 < 0 ? 0 : (y1 > 6 ? 6 : y1);
  int x0c = x0 < 0 ? 0 : (x0 > 6 ? 6 : x0);
  int x1 = x0 + 1; int x1c = x1 < 0 ? 0 : (x1 > 6 ? 6 : x1);
  float a = (1.f - wx) * t7[y0c*7 + x0c] + wx * t7[y0c*7 + x1c];
  float b = (1.f - wx) * t7[y1c*7 + x0c] + wx * t7[y1c*7 + x1c];
  return (1.f - wy) * a + wy * b;
}

__global__ void bias_kernel(const float* __restrict__ an, const float* __restrict__ ah,
                            const float* __restrict__ aw, const float* __restrict__ na,
                            const float* __restrict__ ha, const float* __restrict__ wa,
                            const float* __restrict__ ac, const float* __restrict__ ca,
                            float* __restrict__ pos, float* __restrict__ ab) {
  int i = blockIdx.x * blockDim.x + threadIdx.x;   // over 12*40*197
  if (i >= NHEAD * NAGENT * SEQ) return;
  int n = i % SEQ; int t = i / SEQ; int a = t % NAGENT; int h = t / NAGENT;
  float pv, av;
  if (n == 0) {
    pv = ac[h*NAGENT + a];
    av = ca[h*NAGENT + a];
  } else {
    int p = n - 1, y = p / WINS, x = p % WINS;
    pv = bilerp7(an + (h*NAGENT + a) * 49, y, x)
       + ah[(h*NAGENT + a) * WINS + y] + aw[(h*NAGENT + a) * WINS + x];
    av = bilerp7(na + (h*NAGENT + a) * 49, y, x)
       + ha[(h*WINS + y) * NAGENT + a] + wa[(h*WINS + x) * NAGENT + a];
  }
  pos[(h*NAGENT + a) * SEQ + n] = pv;
  ab[(h*SEQ + n) * NAGENT + a] = av;
}

// ---------------------------------------------------------------------------
// agent tokens: ta_attn = softmax(ta_q . keys^T * 0.125); agent = ta_attn . x_img
// one block per batch. M pad 48, n pad 224. Scores live in LDS as f32.
// ---------------------------------------------------------------------------
__global__ void __launch_bounds__(256)
agent_kernel(const float* __restrict__ x,        // [B][197][768]
             const float* __restrict__ keys,     // [MTOK][64] rows b*197+n
             const float* __restrict__ ta_q,     // [40][64]
             _Float16* __restrict__ agenth) {    // [B][40][768]
  __shared__ float    S[48 * 224];               // 43008 B
  __shared__ _Float16 aq[48][64];                // 6144 B
  const int b = blockIdx.x;
  const int tid = threadIdx.x, lane = tid & 31, wid = tid >> 5;
  const int half = lane >> 4, lr = lane & 15;

  for (int i = tid; i < 48 * 64; i += 256) {
    int r = i >> 6, c = i & 63;
    aq[r][c] = (r < NAGENT) ? (_Float16)ta_q[r * 64 + c] : (_Float16)0.f;
  }
  __syncthreads();

  // scores S[a][n], 3 x 14 tiles, K = 64 (2 WMMA steps)
  for (int t = wid; t < 3 * 14; t += 8) {
    int tm = t / 14, tn = t % 14;
    v8f acc; for (int r = 0; r < 8; ++r) acc[r] = 0.f;
    int n = tn * 16 + lr;
    for (int ks = 0; ks < 2; ++ks) {
      int k0 = ks * 32;
      v16h af = pack_a_h(&aq[tm*16 + lr][0], k0, half);
      v16h bf;
      if (n < SEQ - 1) {
        const float* kp = keys + ((size_t)(b*SEQ + 1 + n)) * 64 + k0 + half*16;
        for (int e = 0; e < 16; ++e) bf[e] = (_Float16)kp[e];
      } else bf = zero_frag();
      acc = wmma16(af, bf, acc);
    }
    for (int r = 0; r < 8; ++r)
      S[(tm*16 + half*8 + r) * 224 + n] = acc[r] * 0.125f;
  }
  __syncthreads();

  // softmax rows a < 40 over n < 196 (in place, normalized f32 probs)
  if (tid < NAGENT) {
    float* row = &S[tid * 224];
    float mx = -1e30f;
    for (int n = 0; n < SEQ - 1; ++n) mx = fmaxf(mx, row[n]);
    float sum = 0.f;
    for (int n = 0; n < SEQ - 1; ++n) { float e = __expf(row[n] - mx); row[n] = e; sum += e; }
    float inv = 1.f / sum;
    for (int n = 0; n < SEQ - 1; ++n) row[n] *= inv;
  }
  __syncthreads();

  // agent[a][c] = sum_n att[a][n] * x[b][1+n][c]; 3 x 48 tiles, K = 224 (7 steps)
  for (int t = wid; t < 3 * 48; t += 8) {
    int tm = t / 48, tc = t % 48;
    int c = tc * 16 + lr;
    v8f acc; for (int r = 0; r < 8; ++r) acc[r] = 0.f;
    for (int ks = 0; ks < 7; ++ks) {
      int k0 = ks * 32;
      v16h af = pack_a_f32(&S[(tm*16 + lr) * 224], k0, half);
      v16h bf;   // NN: bf[e] = x[b][1 + k0+half*16+e][c]
      for (int e = 0; e < 16; ++e) {
        int n = k0 + half*16 + e;
        bf[e] = (n < SEQ - 1) ? (_Float16)x[((size_t)b*SEQ + 1 + n) * DIMC + c]
                              : (_Float16)0.f;
      }
      acc = wmma16(af, bf, acc);
    }
    for (int r = 0; r < 8; ++r) {
      int a = tm*16 + half*8 + r;
      if (a < NAGENT) agenth[((size_t)b*NAGENT + a) * DIMC + c] = (_Float16)acc[r];
    }
  }
}

// ---------------------------------------------------------------------------
// agent attention core, one block per (head, batch)
// ---------------------------------------------------------------------------
__global__ void __launch_bounds__(256)
attn_kernel(const _Float16* __restrict__ qkvh,   // [MTOK][2304] (q|k|v)
            const _Float16* __restrict__ agenth, // [B][40][768]
            const float* __restrict__ pos,       // [12][40][197]
            const float* __restrict__ ab,        // [12][197][40]
            float* __restrict__ attn_out) {      // [MTOK][768]
  __shared__ float    S[48 * 224];               // scores (reused), 43008 B
  __shared__ _Float16 Ag[48][64];                // agent (padded), 6144 B
  __shared__ _Float16 AgV[48][64];               // agent_v, 6144 B
  const int h = blockIdx.x, b = blockIdx.y;
  const int tid = threadIdx.x, lane = tid & 31, wid = tid >> 5;
  const int half = lane >> 4, lr = lane & 15;
  const size_t qkv_row = (size_t)b * SEQ;

  for (int i = tid; i < 48 * 64; i += 256) {
    int a = i >> 6, d = i & 63;
    Ag[a][d] = (a < NAGENT) ? agenth[((size_t)b*NAGENT + a) * DIMC + h*HDIM + d]
                            : (_Float16)0.f;
  }
  __syncthreads();

  // P2: S1[a][n] = 0.125 * (agent . k^T) + pos ; 3 x 14 tiles, K = 64
  for (int t = wid; t < 3 * 14; t += 8) {
    int tm = t / 14, tn = t % 14;
    int n = tn * 16 + lr;
    v8f acc; for (int r = 0; r < 8; ++r) acc[r] = 0.f;
    for (int ks = 0; ks < 2; ++ks) {
      int k0 = ks * 32;
      v16h af = pack_a_h(&Ag[tm*16 + lr][0], k0, half);
      v16h bf;
      if (n < SEQ) {
        const _Float16* kp = qkvh + (qkv_row + n) * 2304 + DIMC + h*HDIM + k0 + half*16;
        v8h u0 = *(const v8h*)kp;
        v8h u1 = *(const v8h*)(kp + 8);
        for (int e = 0; e < 8; ++e) { bf[e] = u0[e]; bf[8+e] = u1[e]; }
      } else bf = zero_frag();
      acc = wmma16(af, bf, acc);
    }
    for (int r = 0; r < 8; ++r) {
      int a = tm*16 + half*8 + r;
      float v = acc[r] * SCALE;
      if (a < NAGENT && n < SEQ) v += pos[(h*NAGENT + a) * SEQ + n];
      S[a * 224 + n] = v;
    }
  }
  __syncthreads();

  // P3: softmax rows a<40 over n<197
  if (tid < NAGENT) {
    float* row = &S[tid * 224];
    float mx = -1e30f;
    for (int n = 0; n < SEQ; ++n) mx = fmaxf(mx, row[n]);
    float sum = 0.f;
    for (int n = 0; n < SEQ; ++n) { float e = __expf(row[n] - mx); row[n] = e; sum += e; }
    float inv = 1.f / sum;
    for (int n = 0; n < SEQ; ++n) row[n] *= inv;
  }
  __syncthreads();

  // P4: agent_v[a][d] = sum_n att[a][n] v[n][d]; 3 x 4 tiles, K = 224 (7 steps)
  for (int t = wid; t < 3 * 4; t += 8) {
    int tm = t / 4, tn = t % 4;
    int d = tn * 16 + lr;
    v8f acc; for (int r = 0; r < 8; ++r) acc[r] = 0.f;
    for (int ks = 0; ks < 7; ++ks) {
      int k0 = ks * 32;
      v16h af = pack_a_f32(&S[(tm*16 + lr) * 224], k0, half);
      v16h bf;   // NN gather from global v
      for (int e = 0; e < 16; ++e) {
        int n = k0 + half*16 + e;
        bf[e] = (n < SEQ) ? qkvh[(qkv_row + n) * 2304 + 1536 + h*HDIM + d]
                          : (_Float16)0.f;
      }
      acc = wmma16(af, bf, acc);
    }
    for (int r = 0; r < 8; ++r)
      AgV[tm*16 + half*8 + r][d] = (_Float16)acc[r];
  }
  __syncthreads();

  // P5: S2[n][a] = 0.125 * (q . agent^T) + ab ; 13 x 3 tiles, K = 64
  for (int t = wid; t < 13 * 3; t += 8) {
    int tm = t / 3, tn = t % 3;
    int m = tm * 16 + lr;
    int a = tn * 16 + lr;
    v8f acc; for (int r = 0; r < 8; ++r) acc[r] = 0.f;
    for (int ks = 0; ks < 2; ++ks) {
      int k0 = ks * 32;
      v16h af;
      if (m < SEQ) af = pack_a_h(qkvh + (qkv_row + m) * 2304 + h*HDIM, k0, half);
      else         af = zero_frag();
      v16h bf = *(const v16h*)&Ag[a][k0 + half*16];
      acc = wmma16(af, bf, acc);
    }
    for (int r = 0; r < 8; ++r) {
      int n = tm*16 + half*8 + r;
      float v = acc[r] * SCALE;
      if (n < SEQ && a < NAGENT) v += ab[(h*SEQ + n) * NAGENT + a];
      S[n * 48 + a] = v;
    }
  }
  __syncthreads();

  // P6: softmax rows n<197 over a<40
  if (tid < SEQ) {
    float* row = &S[tid * 48];
    float mx = -1e30f;
    for (int a = 0; a < NAGENT; ++a) mx = fmaxf(mx, row[a]);
    float sum = 0.f;
    for (int a = 0; a < NAGENT; ++a) { float e = __expf(row[a] - mx); row[a] = e; sum += e; }
    float inv = 1.f / sum;
    for (int a = 0; a < NAGENT; ++a) row[a] *= inv;
  }
  __syncthreads();

  // P7: out[n][d] = sum_a att2[n][a] agent_v[a][d]; 13 x 4 tiles, K = 64 (guard a<48)
  for (int t = wid; t < 13 * 4; t += 8) {
    int tm = t / 4, tn = t % 4;
    int d = tn * 16 + lr;
    v8f acc; for (int r = 0; r < 8; ++r) acc[r] = 0.f;
    for (int ks = 0; ks < 2; ++ks) {
      int k0 = ks * 32;
      v16h af = pack_a_f32_guard(&S[(tm*16 + lr) * 48], k0, half, 48);
      v16h bf;   // NN from AgV
      for (int e = 0; e < 16; ++e) {
        int a = k0 + half*16 + e;
        bf[e] = (a < 48) ? AgV[a][d] : (_Float16)0.f;
      }
      acc = wmma16(af, bf, acc);
    }
    for (int r = 0; r < 8; ++r) {
      int n = tm*16 + half*8 + r;
      if (n < SEQ)
        attn_out[(qkv_row + n) * DIMC + h*HDIM + d] = acc[r];
    }
  }
}

// ---------------------------------------------------------------------------
// depthwise 3x3 conv on v + residual combine + cast to f16
// ---------------------------------------------------------------------------
__global__ void combine_conv_kernel(const float* __restrict__ attn,
                                    const _Float16* __restrict__ qkvh,
                                    const float* __restrict__ dwc_w,
                                    const float* __restrict__ dwc_b,
                                    _Float16* __restrict__ outh) {
  long long i = (long long)blockIdx.x * blockDim.x + threadIdx.x;
  if (i >= (long long)MTOK * DIMC) return;
  int c = (int)(i % DIMC);
  long long t = i / DIMC;
  int n = (int)(t % SEQ);
  int b = (int)(t / SEQ);
  float v = attn[i];
  if (n > 0) {
    int p = n - 1, y = p / WINS, x = p % WINS;
    float s = dwc_b[c];
    for (int dy = -1; dy <= 1; ++dy)
      for (int dx = -1; dx <= 1; ++dx) {
        int yy = y + dy, xx = x + dx;
        if (yy >= 0 && yy < WINS && xx >= 0 && xx < WINS) {
          float w  = dwc_w[c * 9 + (dy + 1) * 3 + (dx + 1)];
          float vv = (float)qkvh[((size_t)b*SEQ + 1 + yy*WINS + xx) * 2304 + 1536 + c];
          s += w * vv;
        }
      }
    v += s;
  }
  outh[i] = (_Float16)v;
}

// ---------------------------------------------------------------------------
// launcher
// ---------------------------------------------------------------------------
extern "C" void kernel_launch(void* const* d_in, const int* in_sizes, int n_in,
                              void* d_out, int out_size, void* d_ws, size_t ws_size,
                              hipStream_t stream) {
  const float* x       = (const float*)d_in[0];
  const float* qkv_w   = (const float*)d_in[1];
  const float* proj_w  = (const float*)d_in[2];
  const float* proj_b  = (const float*)d_in[3];
  const float* dwc_w   = (const float*)d_in[4];
  const float* dwc_b   = (const float*)d_in[5];
  const float* an_bias = (const float*)d_in[6];
  const float* ah_bias = (const float*)d_in[7];
  const float* aw_bias = (const float*)d_in[8];
  const float* na_bias = (const float*)d_in[9];
  const float* ha_bias = (const float*)d_in[10];
  const float* wa_bias = (const float*)d_in[11];
  const float* ac_bias = (const float*)d_in[12];
  const float* ca_bias = (const float*)d_in[13];
  const float* ta_q    = (const float*)d_in[14];
  const float* ta_kw   = (const float*)d_in[15];
  const float* ta_kb   = (const float*)d_in[16];
  float* out = (float*)d_out;

  // workspace layout (256 B aligned)
  char* ws = (char*)d_ws;
  size_t o = 0;
  auto carve = [&](size_t bytes) { size_t r = o; o = (o + bytes + 255) & ~(size_t)255; return r; };
  size_t off_xh     = carve((size_t)MTOK * DIMC * 2);
  size_t off_qkvwh  = carve((size_t)3 * DIMC * DIMC * 2);
  size_t off_projwh = carve((size_t)DIMC * DIMC * 2);
  size_t off_takwh  = carve((size_t)64 * DIMC * 2);
  size_t off_qkvh   = carve((size_t)MTOK * 3 * DIMC * 2);
  size_t off_keys   = carve((size_t)MTOK * 64 * 4);
  size_t off_agenth = carve((size_t)NB * NAGENT * DIMC * 2);
  size_t off_pos    = carve((size_t)NHEAD * NAGENT * SEQ * 4);
  size_t off_ab     = carve((size_t)NHEAD * SEQ * NAGENT * 4);
  size_t off_attn   = carve((size_t)MTOK * DIMC * 4);
  size_t off_outh   = carve((size_t)MTOK * DIMC * 2);
  (void)ws_size;

  _Float16* xh     = (_Float16*)(ws + off_xh);
  _Float16* qkvwh  = (_Float16*)(ws + off_qkvwh);
  _Float16* projwh = (_Float16*)(ws + off_projwh);
  _Float16* takwh  = (_Float16*)(ws + off_takwh);
  _Float16* qkvh   = (_Float16*)(ws + off_qkvh);
  float*    keys   = (float*)   (ws + off_keys);
  _Float16* agenth = (_Float16*)(ws + off_agenth);
  float*    pos    = (float*)   (ws + off_pos);
  float*    ab     = (float*)   (ws + off_ab);
  float*    attn   = (float*)   (ws + off_attn);
  _Float16* outh   = (_Float16*)(ws + off_outh);

  auto blocks = [](long long n, int bs) { return (unsigned)((n + bs - 1) / bs); };

  // 1) casts to f16
  cast_f16_kernel<<<blocks((long long)MTOK * DIMC, 256), 256, 0, stream>>>(x, xh, (long long)MTOK * DIMC);
  cast_f16_kernel<<<blocks((long long)3 * DIMC * DIMC, 256), 256, 0, stream>>>(qkv_w, qkvwh, (long long)3 * DIMC * DIMC);
  cast_f16_kernel<<<blocks((long long)DIMC * DIMC, 256), 256, 0, stream>>>(proj_w, projwh, (long long)DIMC * DIMC);
  cast_f16_kernel<<<blocks((long long)64 * DIMC, 256), 256, 0, stream>>>(ta_kw, takwh, (long long)64 * DIMC);

  // 2) qkv = x . Wqkv^T  (f16 out), M=25216 N=2304 K=768
  {
    dim3 g((MTOK + 127) / 128, (3 * DIMC + 127) / 128);
    gemm_nt_wmma<true><<<g, 256, 0, stream>>>(xh, qkvwh, nullptr, nullptr, qkvh,
                                              MTOK, 3 * DIMC, DIMC);
  }
  // 3) keys = x . ta_kw^T + ta_kb (f32 out), N=64
  {
    dim3 g((MTOK + 127) / 128, 1);
    gemm_nt_wmma<false><<<g, 256, 0, stream>>>(xh, takwh, ta_kb, keys, nullptr,
                                               MTOK, 64, DIMC);
  }
  // 4) positional biases
  bias_kernel<<<blocks((long long)NHEAD * NAGENT * SEQ, 256), 256, 0, stream>>>(
      an_bias, ah_bias, aw_bias, na_bias, ha_bias, wa_bias, ac_bias, ca_bias, pos, ab);

  // 5) agent tokens
  agent_kernel<<<NB, 256, 0, stream>>>(x, keys, ta_q, agenth);

  // 6) agent attention (per head, batch)
  {
    dim3 g(NHEAD, NB);
    attn_kernel<<<g, 256, 0, stream>>>(qkvh, agenth, pos, ab, attn);
  }

  // 7) depthwise conv + combine + cast
  combine_conv_kernel<<<blocks((long long)MTOK * DIMC, 256), 256, 0, stream>>>(
      attn, qkvh, dwc_w, dwc_b, outh);

  // 8) out = combined . Wproj^T + proj_b (f32 out)
  {
    dim3 g((MTOK + 127) / 128, (DIMC + 127) / 128);
    gemm_nt_wmma<false><<<g, 256, 0, stream>>>(outh, projwh, proj_b, out, nullptr,
                                               MTOK, DIMC, DIMC);
  }
  (void)in_sizes; (void)n_in; (void)out_size;
}
